// LocalFeatureGuided_39221641347601
// MI455X (gfx1250) — matrix operand order repaired and checked by reading, not compiled, not run on hardware
//
#include <hip/hip_runtime.h>
#include <hip/hip_bf16.h>
#include <math.h>

typedef __attribute__((ext_vector_type(16))) _Float16 v16h;
typedef __attribute__((ext_vector_type(8)))  float    v8f;

#define BB 16
#define CC 128
#define HH 128
#define WWD 128
#define H2G 64
#define LL 4096
#define LT 64          // l-positions per block tile (one full h2 row)
#define NTILES 64      // L / LT

union AFrag { v16h v; unsigned int u[8]; };

__device__ __forceinline__ v8f wmma16(v16h a, v16h b, v8f c) {
  return __builtin_amdgcn_wmma_f32_16x16x32_f16(
      /*neg_a=*/false, a, /*neg_b=*/false, b,
      /*c_mod=*/(short)0, c, /*reuse_a=*/false, /*reuse_b=*/false);
}

// A-matrix 16x32 f16 fragment (contiguous k order, not the ISA table order): lane m = lane&15,
// element i -> K = kbase + i + 16*(lane>=16). Built as 8 dword loads.
__device__ __forceinline__ v16h load_a_frag(const _Float16* w, int mbase, int kbase, int lane) {
  AFrag f;
  const int hi = (lane >> 4) & 1;
  const int d  = mbase + (lane & 15);
  const _Float16* base = w + (size_t)d * CC + kbase + hi * 16;
#pragma unroll
  for (int j = 0; j < 8; ++j) {
    int k = 2 * j;
    f.u[j] = *(const unsigned int*)(base + k);
  }
  return f.v;
}

// B-matrix 32x16 f16 swizzle (contiguous k order, same as load_a_frag):
// element j -> K = j + 16*(lane>=16), N = lane&15.
__device__ __forceinline__ void swz_b(int c, int l, int& kk, int& lane, int& i) {
  kk = c >> 5;
  int kb = c & 31;
  int hi = (kb >> 4) & 1;
  i = kb & 15;
  lane = (hi << 4) | (l & 15);
}
// LDS half-index for token tile: ((m*4+kk)*4 + sub)*512 + lane*16 + i  (sub = l>>4)

// ---------------- K0: f32->f16 weight copies + zero S ----------------
__global__ __launch_bounds__(256) void k_init(const float* qkv_w, const float* proj_w,
                                              _Float16* wq_h, _Float16* wp_h, float* S) {
  int i = blockIdx.x * 256 + threadIdx.x;
  if (i < 384 * CC) wq_h[i] = (_Float16)qkv_w[i];
  if (i < CC * CC)  wp_h[i] = (_Float16)proj_w[i];
  if (i < BB * CC * 5) S[i] = 0.0f;
}

// ---------------- K1: guide = dwconv7x7_s2(gelu(bn(x))) ----------------
__global__ __launch_bounds__(256) void k_guide(const float* __restrict__ x,
                                               const float* __restrict__ gamma,
                                               const float* __restrict__ beta,
                                               const float* __restrict__ mean,
                                               const float* __restrict__ var,
                                               const float* __restrict__ dww,
                                               const float* __restrict__ dwb,
                                               float* __restrict__ guide) {
  extern __shared__ float g[];               // 134*136 + 64
  float* wsh = g + 134 * 136;
  const int tid = threadIdx.x;
  const int bc = blockIdx.x;                 // b*C + c
  const int c = bc & (CC - 1);
  const float sc = gamma[c] * rsqrtf(var[c] + 1e-5f);
  const float sh = beta[c] - mean[c] * sc;
  const float* xs = x + (size_t)bc * (HH * WWD);

  if (tid < 49) wsh[tid] = dww[c * 49 + tid];
  for (int idx = tid; idx < 134 * 134; idx += 256) {
    int row = idx / 134, col = idx - row * 134;
    int ih = row - 3, iw = col - 3;
    float v = 0.0f;
    if (ih >= 0 && ih < HH && iw >= 0 && iw < WWD) {
      float xv = xs[ih * WWD + iw] * sc + sh;
      v = 0.5f * xv * (1.0f + erff(xv * 0.70710678118654752f));   // exact GELU
    }
    g[row * 136 + col] = v;
  }
  __syncthreads();

  const float bias = dwb[c];
  for (int o = tid; o < LL; o += 256) {
    int oh = o >> 6, ow = o & 63;
    float acc = bias;
#pragma unroll
    for (int ky = 0; ky < 7; ++ky)
#pragma unroll
      for (int kx = 0; kx < 7; ++kx)
        acc += g[(2 * oh + ky) * 136 + 2 * ow + kx] * wsh[ky * 7 + kx];
    guide[(size_t)bc * LL + o] = acc;
  }
}

// -------- token staging into B-swizzled LDS (shared by K2 / K4) --------
__device__ __forceinline__ void stage_tokens(_Float16* tokB, const float* __restrict__ x,
                                             const float* __restrict__ guide,
                                             int b, int h2, int tid) {
  // m = 0: guide row
  const float* gsrc = guide + (size_t)(b * CC) * LL + h2 * 64;
  for (int idx = tid; idx < CC * LT; idx += 256) {
    int c = idx >> 6, l = idx & 63;
    int kk, ln, i; swz_b(c, l, kk, ln, i);
    tokB[((kk * 4) + (l >> 4)) * 512 + ln * 16 + i] = (_Float16)gsrc[(size_t)c * LL + l];
  }
  // m = 1..4: 2x2 window pixels of x
#pragma unroll
  for (int m = 1; m <= 4; ++m) {
    int wsh = (m - 1) >> 1, wsw = (m - 1) & 1;
    const float* xsrc = x + ((size_t)(b * CC) * HH + (2 * h2 + wsh)) * WWD + wsw;
    for (int idx = tid; idx < CC * LT; idx += 256) {
      int c = idx >> 6, l = idx & 63;
      int kk, ln, i; swz_b(c, l, kk, ln, i);
      tokB[((m * 4 + kk) * 4 + (l >> 4)) * 512 + ln * 16 + i] =
          (_Float16)xsrc[(size_t)c * HH * WWD + 2 * l];
    }
  }
}

// ---------------- K2: q0 & k via WMMA, partial attention scores ----------------
__global__ __launch_bounds__(256) void k_qk(const float* __restrict__ x,
                                            const float* __restrict__ guide,
                                            const _Float16* __restrict__ wq_h,
                                            const float* __restrict__ qkv_b,
                                            float* __restrict__ S) {
  extern __shared__ char smem[];
  _Float16* tokB = (_Float16*)smem;          // 5*4*4*512 halfs = 81920 B
  const int tid = threadIdx.x;
  const int lane = tid & 31, w = tid >> 5;
  const int b = blockIdx.x >> 6, h2 = blockIdx.x & 63;

  stage_tokens(tokB, x, guide, b, h2, tid);
  __syncthreads();

  const int hi = lane >> 4, nl = lane & 15;
  float qb[8], kb[8];
#pragma unroll
  for (int r = 0; r < 8; ++r) {
    int c = 16 * w + r + 8 * hi;
    qb[r] = qkv_b[c];
    kb[r] = qkv_b[CC + c];
  }

  // q at the guide token (m=0), rows d = 16w..16w+15 (+8*hi in D layout)
  v8f q0[4];
  {
    v16h aq[4];
#pragma unroll
    for (int kk = 0; kk < 4; ++kk) aq[kk] = load_a_frag(wq_h, 16 * w, 32 * kk, lane);
#pragma unroll
    for (int sub = 0; sub < 4; ++sub) {
      v8f acc = {};
#pragma unroll
      for (int kk = 0; kk < 4; ++kk) {
        v16h bf = *(const v16h*)&tokB[((0 * 4 + kk) * 4 + sub) * 512 + lane * 16];
        acc = wmma16(aq[kk], bf, acc);
      }
#pragma unroll
      for (int r = 0; r < 8; ++r) acc[r] += qb[r];
      q0[sub] = acc;
    }
  }

  // k for each of the 5 tokens; accumulate partial S[b,c,m] = sum_l q0*k
  v16h ak[4];
#pragma unroll
  for (int kk = 0; kk < 4; ++kk) ak[kk] = load_a_frag(wq_h, CC + 16 * w, 32 * kk, lane);
#pragma unroll
  for (int m = 0; m < 5; ++m) {
    float s[8];
#pragma unroll
    for (int r = 0; r < 8; ++r) s[r] = 0.0f;
#pragma unroll
    for (int sub = 0; sub < 4; ++sub) {
      v8f acc = {};
#pragma unroll
      for (int kk = 0; kk < 4; ++kk) {
        v16h bf = *(const v16h*)&tokB[((m * 4 + kk) * 4 + sub) * 512 + lane * 16];
        acc = wmma16(ak[kk], bf, acc);
      }
#pragma unroll
      for (int r = 0; r < 8; ++r) s[r] += q0[sub][r] * (acc[r] + kb[r]);
    }
#pragma unroll
    for (int r = 0; r < 8; ++r) {     // reduce over the 16 n-lanes of each half
      float v = s[r];
      v += __shfl_xor(v, 1, 32);
      v += __shfl_xor(v, 2, 32);
      v += __shfl_xor(v, 4, 32);
      v += __shfl_xor(v, 8, 32);
      if (nl == 0) {
        int c = 16 * w + r + 8 * hi;
        atomicAdd(&S[(b * CC + c) * 5 + m], v);
      }
    }
  }
}

// ---------------- K3: per-(b,c) 5-way softmax ----------------
__global__ __launch_bounds__(256) void k_softmax(const float* __restrict__ S,
                                                 float* __restrict__ P) {
  int i = blockIdx.x * 256 + threadIdx.x;
  if (i >= BB * CC) return;
  float s[5], mx = -3.4e38f;
#pragma unroll
  for (int m = 0; m < 5; ++m) { s[m] = S[i * 5 + m] * 0.08838834764831845f; mx = fmaxf(mx, s[m]); }
  float sum = 0.0f;
#pragma unroll
  for (int m = 0; m < 5; ++m) { s[m] = expf(s[m] - mx); sum += s[m]; }
  float inv = 1.0f / sum;
#pragma unroll
  for (int m = 0; m < 5; ++m) P[i * 5 + m] = s[m] * inv;
}

// ---------------- K4: v via WMMA, attention combine, fused proj ----------------
__global__ __launch_bounds__(256) void k_out(const float* __restrict__ x,
                                             const float* __restrict__ guide,
                                             const _Float16* __restrict__ wq_h,
                                             const _Float16* __restrict__ wp_h,
                                             const float* __restrict__ qkv_b,
                                             const float* __restrict__ proj_b,
                                             const float* __restrict__ P,
                                             float* __restrict__ out) {
  extern __shared__ char smem[];
  _Float16* tokB = (_Float16*)smem;               // 81920 B
  float* Pl = (float*)(smem + 81920);             // 640 floats
  const int tid = threadIdx.x;
  const int lane = tid & 31, w = tid >> 5;
  const int b = blockIdx.x >> 6, h2 = blockIdx.x & 63;
  const int lbase = h2 * 64;

  stage_tokens(tokB, x, guide, b, h2, tid);
  for (int idx = tid; idx < CC * 5; idx += 256) Pl[idx] = P[b * CC * 5 + idx];
  __syncthreads();

  const int hi = lane >> 4, nl = lane & 15;

  // o[c,l] = sum_m P[c,m] * (Wv . token_m)[c,l]  (+ v_bias, since sum_m P = 1)
  v16h av[4];
#pragma unroll
  for (int kk = 0; kk < 4; ++kk) av[kk] = load_a_frag(wq_h, 2 * CC + 16 * w, 32 * kk, lane);
  v8f o[4];
#pragma unroll
  for (int sub = 0; sub < 4; ++sub) { v8f z = {}; o[sub] = z; }
#pragma unroll
  for (int m = 0; m < 5; ++m) {
#pragma unroll
    for (int sub = 0; sub < 4; ++sub) {
      v8f acc = {};
#pragma unroll
      for (int kk = 0; kk < 4; ++kk) {
        v16h bf = *(const v16h*)&tokB[((m * 4 + kk) * 4 + sub) * 512 + lane * 16];
        acc = wmma16(av[kk], bf, acc);
      }
#pragma unroll
      for (int r = 0; r < 8; ++r) {
        int c = 16 * w + r + 8 * hi;
        o[sub][r] += Pl[c * 5 + m] * acc[r];
      }
    }
  }
#pragma unroll
  for (int sub = 0; sub < 4; ++sub)
#pragma unroll
    for (int r = 0; r < 8; ++r) o[sub][r] += qkv_b[2 * CC + 16 * w + r + 8 * hi];

  __syncthreads();   // done reading tokens; reuse LDS for the proj operand
#pragma unroll
  for (int sub = 0; sub < 4; ++sub)
#pragma unroll
    for (int r = 0; r < 8; ++r) {
      int c = 16 * w + r + 8 * hi;
      int l = 16 * sub + nl;
      int kk, ln, i; swz_b(c, l, kk, ln, i);
      tokB[(kk * 4 + sub) * 512 + ln * 16 + i] = (_Float16)o[sub][r];
    }
  __syncthreads();

  // out[d,l] = proj_w[d,:] . o[:,l] + proj_b[d]
  v16h ap[4];
#pragma unroll
  for (int kk = 0; kk < 4; ++kk) ap[kk] = load_a_frag(wp_h, 16 * w, 32 * kk, lane);
  float pb[8];
#pragma unroll
  for (int r = 0; r < 8; ++r) pb[r] = proj_b[16 * w + r + 8 * hi];
#pragma unroll
  for (int sub = 0; sub < 4; ++sub) {
    v8f acc = {};
#pragma unroll
    for (int kk = 0; kk < 4; ++kk) {
      v16h bf = *(const v16h*)&tokB[(kk * 4 + sub) * 512 + lane * 16];
      acc = wmma16(ap[kk], bf, acc);
    }
#pragma unroll
    for (int r = 0; r < 8; ++r) {
      int d = 16 * w + r + 8 * hi;
      out[(size_t)(b * CC + d) * LL + lbase + 16 * sub + nl] = acc[r] + pb[r];
    }
  }
}

extern "C" void kernel_launch(void* const* d_in, const int* in_sizes, int n_in,
                              void* d_out, int out_size, void* d_ws, size_t ws_size,
                              hipStream_t stream) {
  const float* x      = (const float*)d_in[0];
  const float* gamma  = (const float*)d_in[1];
  const float* beta   = (const float*)d_in[2];
  const float* mean   = (const float*)d_in[3];
  const float* var    = (const float*)d_in[4];
  const float* dww    = (const float*)d_in[5];
  const float* dwb    = (const float*)d_in[6];
  const float* qkv_w  = (const float*)d_in[7];
  const float* qkv_b  = (const float*)d_in[8];
  const float* proj_w = (const float*)d_in[9];
  const float* proj_b = (const float*)d_in[10];
  float* out = (float*)d_out;

  char* ws = (char*)d_ws;
  float*     guide = (float*)ws;                              // 16*128*4096 f32 = 32 MiB
  _Float16*  wq_h  = (_Float16*)(ws + 33554432);              // 384*128 f16
  _Float16*  wp_h  = (_Float16*)(ws + 33554432 + 98304);      // 128*128 f16
  float*     S     = (float*)(ws + 33554432 + 98304 + 32768); // 16*128*5 f32
  float*     P     = (float*)(ws + 33554432 + 98304 + 32768 + 40960);

  k_init<<<192, 256, 0, stream>>>(qkv_w, proj_w, wq_h, wp_h, S);
  k_guide<<<BB * CC, 256, (134 * 136 + 64) * sizeof(float), stream>>>(
      x, gamma, beta, mean, var, dww, dwb, guide);
  k_qk<<<BB * NTILES, 256, 81920, stream>>>(x, guide, wq_h, qkv_b, S);
  k_softmax<<<(BB * CC + 255) / 256, 256, 0, stream>>>(S, P);
  k_out<<<BB * NTILES, 256, 81920 + 2560 + 64, stream>>>(
      x, guide, wq_h, wp_h, qkv_b, proj_b, P, out);
}
